// GraphSAGE_DGL_82609400971391
// MI455X (gfx1250) — matrix operand order repaired and checked by reading, not compiled, not run on hardware
//
#include <hip/hip_runtime.h>
#include <hip/hip_bf16.h>
#include <stdint.h>

typedef __attribute__((ext_vector_type(16))) __bf16 v16bf;
typedef __attribute__((ext_vector_type(8)))  __bf16 v8bf;
typedef __attribute__((ext_vector_type(8)))  float  v8f;

#define DD 128      // feature dim
#define RR 3        // relations
#define KTOT 512    // folded K = 4*DD  ( [h | mean0 | mean1 | mean2] )

static __device__ __forceinline__ unsigned short f2bf(float f) {
    union { float f; unsigned int u; } c; c.f = f;
    unsigned int r = ((c.u >> 16) & 1u) + 0x7FFFu;   // round to nearest even
    return (unsigned short)((c.u + r) >> 16);
}

// ---------------- zero fill (float4) ----------------
__global__ void gs_zero4(float4* __restrict__ p, int n4) {
    int i = blockIdx.x * blockDim.x + threadIdx.x;
    if (i < n4) p[i] = make_float4(0.f, 0.f, 0.f, 0.f);
}

// ---------------- degree: one thread per (r,edge) ----------------
__global__ void gs_degree(const int* __restrict__ dst, float* __restrict__ deg,
                          int E, int N, int total) {
    int i = blockIdx.x * blockDim.x + threadIdx.x;
    if (i >= total) return;
    int r = i / E;
    int d = dst[i];
    atomicAdd(deg + (size_t)r * N + d, 1.0f);
}

// ---------------- scatter-add messages: thread = (r,edge,chunk-of-4) ----------------
__global__ void gs_scatter(const float* __restrict__ h, const int* __restrict__ src,
                           const int* __restrict__ dst, float* __restrict__ agg,
                           int E, long long total) {
    long long idx = (long long)blockIdx.x * blockDim.x + threadIdx.x;
    if (idx >= total) return;
    int c = (int)(idx & 31);          // 32 chunks x 4 floats = 128
    long long ea = idx >> 5;          // r*E + e
    int r = (int)(ea / E);
    int s = src[ea];
    int d = dst[ea];
    const float4 v = ((const float4*)(h + (size_t)s * DD))[c];
    float* a = agg + (size_t)d * (RR * DD) + r * DD + c * 4;
    atomicAdd(a + 0, v.x);
    atomicAdd(a + 1, v.y);
    atomicAdd(a + 2, v.z);
    atomicAdd(a + 3, v.w);
}

// ---------------- build X = [h | mean0 | mean1 | mean2] in bf16 ----------------
__global__ void gs_finalize(const float* __restrict__ h, const float* __restrict__ agg,
                            const float* __restrict__ deg, unsigned short* __restrict__ Xb,
                            int N, int total) {
    int i = blockIdx.x * blockDim.x + threadIdx.x;
    if (i >= total) return;
    int node = i >> 9;                // /512
    int col  = i & (KTOT - 1);
    float v;
    if (col < DD) {
        v = h[(size_t)node * DD + col];
    } else {
        int rc = col - DD;            // r*128 + c
        int r  = rc >> 7;
        v = agg[(size_t)node * (RR * DD) + rc] / fmaxf(deg[(size_t)r * N + node], 1.0f);
    }
    Xb[i] = f2bf(v);
}

// ---------------- fold weights: WcatT[n][k], k<128: W_self@SumWm ; else W_neigh@Wm_r ----
__global__ void gs_wprep(const float* __restrict__ W_self, const float* __restrict__ W_neigh,
                         const float* __restrict__ Wm, unsigned short* __restrict__ WcatT,
                         int layer) {
    int i = blockIdx.x * blockDim.x + threadIdx.x;   // 512*128 threads
    if (i >= KTOT * DD) return;
    int kk = i >> 7;          // 0..511
    int n  = i & (DD - 1);
    const float* Wm_l = Wm + (size_t)layer * (RR * DD) * DD;
    float s = 0.f;
    if (kk < DD) {
        const float* ws = W_self + (size_t)layer * DD * DD + (size_t)kk * DD;
        for (int j = 0; j < DD; ++j) {
            float wm = Wm_l[(size_t)j * DD + n] + Wm_l[(size_t)(DD + j) * DD + n]
                     + Wm_l[(size_t)(2 * DD + j) * DD + n];
            s += ws[j] * wm;
        }
    } else {
        int r = (kk >> 7) - 1;
        int k = kk & (DD - 1);
        const float* wn = W_neigh + (size_t)layer * DD * DD + (size_t)k * DD;
        const float* wmr = Wm_l + (size_t)r * DD * DD;
        for (int j = 0; j < DD; ++j)
            s += wn[j] * wmr[(size_t)j * DD + n];
    }
    WcatT[(size_t)n * KTOT + kk] = f2bf(s);
}

__global__ void gs_bias(const float* __restrict__ b_self, const float* __restrict__ Wm,
                        const float* __restrict__ bm, float* __restrict__ bc, int layer) {
    int n = threadIdx.x;
    if (n >= DD) return;
    const float* Wm_l = Wm + (size_t)layer * (RR * DD) * DD;
    float s = bm[(size_t)layer * DD + n];
    for (int j = 0; j < DD; ++j) {
        float wm = Wm_l[(size_t)j * DD + n] + Wm_l[(size_t)(DD + j) * DD + n]
                 + Wm_l[(size_t)(2 * DD + j) * DD + n];
        s += b_self[(size_t)layer * DD + j] * wm;
    }
    bc[n] = s;
}

// ---------------- WMMA GEMM:  out[N x 128] = Xb[N x 512] @ Wcat + bc (opt relu) --------
// Block = 256 threads = 8 waves; block tile = 128 rows; wave w owns rows [w*16, w*16+16).
// B (WcatT, N-major 128x512 bf16) staged in LDS 32KB chunks (128 N x 128 K).
__global__ __launch_bounds__(256)
void gs_gemm(const unsigned short* __restrict__ Xb, const unsigned short* __restrict__ WcatT,
             const float* __restrict__ bc, float* __restrict__ out, int n_rows, int relu) {
    __shared__ unsigned short lds_b[DD * DD];   // 128 x 128 bf16 = 32 KB

    const int tid  = threadIdx.x;
    const int wave = tid >> 5;
    const int lane = tid & 31;
    const int lgrp = lane >> 4;                 // 0: lanes 0-15, 1: lanes 16-31
    const int lmod = lane & 15;

    const int m0 = blockIdx.x * 128 + wave * 16;
    int am = m0 + lmod;
    if (am >= n_rows) am = n_rows - 1;          // clamp: keep EXEC all-ones for WMMA
    const unsigned short* arow = Xb + (size_t)am * KTOT;

    v8f acc[8];
#pragma unroll
    for (int t = 0; t < 8; ++t) acc[t] = (v8f)(0.0f);

    for (int kc = 0; kc < KTOT / DD; ++kc) {    // 4 chunks of K=128
        __syncthreads();
        // stage WcatT[:, kc*128 : kc*128+128] into LDS (N-major, 16B vectors)
        for (int g = tid; g < (DD * DD) / 8; g += 256) {
            int n  = g >> 4;
            int k8 = (g & 15) << 3;
            *(uint4*)(lds_b + (n << 7) + k8) =
                *(const uint4*)(WcatT + ((size_t)n << 9) + kc * DD + k8);
        }
        __syncthreads();

#pragma unroll
        for (int ktl = 0; ktl < 4; ++ktl) {     // K = 32 per WMMA
            const int kbase = kc * DD + ktl * 32;
            // ---- A fragment (contiguous K order, same as B): lane grp 0 -> K {0..15}, grp1 -> {16..31}
            const unsigned short* ap = arow + kbase + (lgrp << 4);
            v8bf lo = *(const v8bf*)(ap);
            v8bf hi = *(const v8bf*)(ap + 8);
            v16bf a;
#pragma unroll
            for (int e = 0; e < 8; ++e) { a[e] = lo[e]; a[8 + e] = hi[e]; }

#pragma unroll
            for (int nt = 0; nt < 8; ++nt) {
                // ---- B fragment: lane -> N = nt*16+lmod; elements = 16 consecutive K,
                //      lanes 16-31 take K+16 (B layout: K striped across lane groups)
                const unsigned short* bp =
                    lds_b + ((nt * 16 + lmod) << 7) + (ktl * 32) + (lgrp << 4);
                v16bf b = *(const v16bf*)bp;
                acc[nt] = __builtin_amdgcn_wmma_f32_16x16x32_bf16(
                    false, a, false, b, (short)0, acc[nt], false, false);
            }
        }
    }

    // ---- store: C layout: lane grp adds +8 to M; VGPR v = M offset; N = lmod
#pragma unroll
    for (int nt = 0; nt < 8; ++nt) {
        const int n = nt * 16 + lmod;
        const float bv = bc[n];
#pragma unroll
        for (int v = 0; v < 8; ++v) {
            const int m = m0 + v + (lgrp << 3);
            if (m < n_rows) {
                float x = acc[nt][v] + bv;
                if (relu) x = fmaxf(x, 0.0f);
                out[(size_t)m * DD + n] = x;
            }
        }
    }
}

// =======================================================================================
extern "C" void kernel_launch(void* const* d_in, const int* in_sizes, int n_in,
                              void* d_out, int out_size, void* d_ws, size_t ws_size,
                              hipStream_t stream) {
    const float* feats   = (const float*)d_in[0];
    const int*   src     = (const int*)d_in[1];
    const int*   dst     = (const int*)d_in[2];
    const float* W_self  = (const float*)d_in[3];
    const float* b_self  = (const float*)d_in[4];
    const float* W_neigh = (const float*)d_in[5];
    const float* Wm      = (const float*)d_in[6];
    const float* bm      = (const float*)d_in[7];
    float* outp = (float*)d_out;

    const int N = in_sizes[0] / DD;        // 50000
    const int E = in_sizes[1] / RR;        // 800000
    const int L = 2;

    // ---- workspace layout (256B aligned) ----
    char* ws = (char*)d_ws;
    size_t off = 0;
    auto take = [&](size_t bytes) { char* p = ws + off; off = (off + bytes + 255) & ~(size_t)255; return p; };
    float*          agg   = (float*)take((size_t)N * RR * DD * sizeof(float));   // 76.8 MB
    float*          deg   = (float*)take((size_t)RR * N * sizeof(float));        // 0.6 MB
    float*          hf    = (float*)take((size_t)N * DD * sizeof(float));        // 25.6 MB
    unsigned short* Xb    = (unsigned short*)take((size_t)N * KTOT * 2);         // 51.2 MB
    unsigned short* WcatT = (unsigned short*)take((size_t)DD * KTOT * 2);        // 128 KB
    float*          bc    = (float*)take(DD * sizeof(float));
    (void)ws_size; (void)n_in; (void)out_size;

    const int TB = 256;

    // degrees (layer-invariant): zero + count
    {
        int n4 = (RR * N + 3) / 4;
        gs_zero4<<<(n4 + TB - 1) / TB, TB, 0, stream>>>((float4*)deg, n4);
        int tot = RR * E;
        gs_degree<<<(tot + TB - 1) / TB, TB, 0, stream>>>(dst, deg, E, N, tot);
    }

    const float* h_in = feats;
    for (int l = 0; l < L; ++l) {
        // zero aggregation buffer
        int n4 = (N * RR * DD) / 4;
        gs_zero4<<<(n4 + TB - 1) / TB, TB, 0, stream>>>((float4*)agg, n4);

        // gather + scatter-add over all relations
        long long tot = (long long)RR * E * 32;
        gs_scatter<<<(unsigned)((tot + TB - 1) / TB), TB, 0, stream>>>(h_in, src, dst, agg, E, tot);

        // fold weights for this layer
        gs_wprep<<<(KTOT * DD + TB - 1) / TB, TB, 0, stream>>>(W_self, W_neigh, Wm, WcatT, l);
        gs_bias<<<1, DD, 0, stream>>>(b_self, Wm, bm, bc, l);

        // X = [h | mean0 | mean1 | mean2] -> bf16
        int ftot = N * KTOT;
        gs_finalize<<<(ftot + TB - 1) / TB, TB, 0, stream>>>(h_in, agg, deg, Xb, N, ftot);

        // one fused WMMA GEMM: h_next = X @ Wcat + bc (+relu for hidden layer)
        float* dstbuf = (l == L - 1) ? outp : hf;
        int relu = (l != L - 1) ? 1 : 0;
        gs_gemm<<<(N + 127) / 128, TB, 0, stream>>>(Xb, WcatT, bc, dstbuf, N, relu);

        h_in = hf;
    }
}
